// GPT_54305566491029
// MI455X (gfx1250) — matrix operand order repaired and checked by reading, not compiled, not run on hardware
//
#include <hip/hip_runtime.h>
#include <hip/hip_bf16.h>

typedef __attribute__((ext_vector_type(16))) __bf16 v16bf;
typedef __attribute__((ext_vector_type(8)))  float  v8f;

#define L_      8
#define T_      2048
#define DIM_    1024
#define NH_     8
#define NKV_    4
#define HD_     128
#define KVD_    512
#define WIN_    384
#define SCALE_  0.12f
#define GATECH_ 32
#define EPS_    1.1920929e-07f

// ---------------------------------------------------------------- utilities

union Frag16 {
  v16bf v;
  uint4 q[2];
};

// Build a 16-element bf16 fragment from two contiguous 16-byte chunks.
__device__ __forceinline__ v16bf ld_pair(const void* p0, const void* p1) {
  Frag16 f;
  f.q[0] = *(const uint4*)p0;
  f.q[1] = *(const uint4*)p1;
  return f.v;
}

// CDNA5 async global->LDS copy (ASYNCcnt-tracked, no VGPR round trip).
__device__ __forceinline__ void async_b128(unsigned lds_off, const __bf16* src) {
  asm volatile("global_load_async_to_lds_b128 %0, %1, off" ::"v"(lds_off), "v"(src)
               : "memory");
}

__device__ __forceinline__ float hmax16(float v) {
  v = fmaxf(v, __shfl_xor(v, 1, 32));
  v = fmaxf(v, __shfl_xor(v, 2, 32));
  v = fmaxf(v, __shfl_xor(v, 4, 32));
  v = fmaxf(v, __shfl_xor(v, 8, 32));
  return v;
}
__device__ __forceinline__ float hsum16(float v) {
  v += __shfl_xor(v, 1, 32);
  v += __shfl_xor(v, 2, 32);
  v += __shfl_xor(v, 4, 32);
  v += __shfl_xor(v, 8, 32);
  return v;
}

// ---------------------------------------------------------------- rmsnorm

__global__ __launch_bounds__(256) void k_rmsnorm(const float* __restrict__ x,
                                                 float* __restrict__ outF,
                                                 __bf16* __restrict__ outB) {
  const int row = blockIdx.x;
  const int tid = threadIdx.x;
  __shared__ float red[256];
  __shared__ float rsh;
  float ss = 0.0f;
  for (int i = tid; i < DIM_; i += 256) {
    float v = x[(size_t)row * DIM_ + i];
    ss += v * v;
  }
  red[tid] = ss;
  __syncthreads();
  for (int s = 128; s > 0; s >>= 1) {
    if (tid < s) red[tid] += red[tid + s];
    __syncthreads();
  }
  if (tid == 0) rsh = rsqrtf(red[0] / (float)DIM_ + EPS_);
  __syncthreads();
  for (int i = tid; i < DIM_; i += 256) {
    float v = x[(size_t)row * DIM_ + i] * rsh;
    outF[(size_t)row * DIM_ + i] = v;
    outB[(size_t)row * DIM_ + i] = (__bf16)v;
  }
}

// ---------------------------------------------------------------- doc ids

__global__ void k_doc(const int* __restrict__ seqlens, int* __restrict__ doc) {
  int p = blockIdx.x * 256 + threadIdx.x;
  if (p < T_) {
    int d = -1;
    for (int i = 0; i < 5; ++i)
      if (seqlens[i] <= p) d++;
    doc[p] = d;
  }
}

// ---------------------------------------------------------------- weight conversion
// f32 [K][N] row-major  ->  bf16 fragment-native layout:
// Wt[((ntile*(K/32) + ktile)*32 + lane)*16 + j]  where
//   k = ktile*32 + (lane>>4)*16 + j ,  n = ntile*16 + (lane&15)
// i.e. each 1KB chunk is one 32x16 WMMA B fragment tile in contiguous k order (matches A).

__global__ __launch_bounds__(256) void k_convW(const float* __restrict__ W,
                                               __bf16* __restrict__ Wt, int K, int N) {
  int idx = blockIdx.x * 256 + threadIdx.x;
  if (idx >= K * N) return;
  int j = idx & 15;
  int lane = (idx >> 4) & 31;
  int t2 = idx >> 9;
  int nkt = K >> 5;
  int ktile = t2 % nkt;
  int ntile = t2 / nkt;
  int k = (ktile << 5) + ((lane >> 4) << 4) + j;
  int n = (ntile << 4) + (lane & 15);
  Wt[idx] = (__bf16)W[(size_t)k * N + n];
}

// ---------------------------------------------------------------- gate + value-embedding

__global__ __launch_bounds__(128) void k_gate(const float* __restrict__ xn,
                                              const float* __restrict__ Wg,
                                              const float* __restrict__ vel,
                                              float* __restrict__ v) {
  const int t = blockIdx.x;
  const int tid = threadIdx.x;
  __shared__ float gs[NKV_];
  int kv = tid >> 5;
  int c = tid & 31;
  float p = xn[(size_t)t * DIM_ + c] * Wg[c * NKV_ + kv];
  for (int m = 16; m >= 1; m >>= 1) p += __shfl_xor(p, m, 32);
  if (c == 0) gs[kv] = 2.0f / (1.0f + __expf(-p));
  __syncthreads();
  for (int k2 = 0; k2 < NKV_; ++k2) {
    size_t idx = (size_t)t * KVD_ + k2 * HD_ + tid;
    v[idx] += gs[k2] * vel[idx];
  }
}

// ---------------------------------------------------------------- rope + head rmsnorm -> bf16
// V is written TRANSPOSED: vT[(kv*HD + hd) * T + t]

__global__ __launch_bounds__(128) void k_rope(const float* __restrict__ q,
                                              const float* __restrict__ k,
                                              const float* __restrict__ v,
                                              const float* __restrict__ cosb,
                                              const float* __restrict__ sinb,
                                              __bf16* __restrict__ qB,
                                              __bf16* __restrict__ kB,
                                              __bf16* __restrict__ vT) {
  const int t = blockIdx.x;
  const int hidx = blockIdx.y;
  const int i = threadIdx.x;
  __shared__ float red[128];
  __shared__ float rsh;

  if (hidx < NH_ + NKV_) {
    const float* src;
    __bf16* dst;
    size_t off;
    if (hidx < NH_) {
      src = q; dst = qB; off = (size_t)t * (NH_ * HD_) + hidx * HD_;
    } else {
      int kv = hidx - NH_;
      src = k; dst = kB; off = (size_t)t * KVD_ + kv * HD_;
    }
    float val;
    if (i < 64) {
      float x1 = src[off + i], x2 = src[off + i + 64];
      val = x1 * cosb[t * 64 + i] + x2 * sinb[t * 64 + i];
    } else {
      int j = i - 64;
      float x1 = src[off + j], x2 = src[off + i];
      val = -x1 * sinb[t * 64 + j] + x2 * cosb[t * 64 + j];
    }
    red[i] = val * val;
    __syncthreads();
    for (int s = 64; s > 0; s >>= 1) {
      if (i < s) red[i] += red[i + s];
      __syncthreads();
    }
    if (i == 0) rsh = rsqrtf(red[0] / (float)HD_ + EPS_);
    __syncthreads();
    dst[off + i] = (__bf16)(val * rsh);
  } else {
    int kv = hidx - NH_ - NKV_;
    vT[((size_t)kv * HD_ + i) * T_ + t] = (__bf16)v[(size_t)t * KVD_ + kv * HD_ + i];
  }
}

// ---------------------------------------------------------------- WMMA GEMM (128x64 tile, 8 waves)
// Barrier-free, async double-buffered: each wave DMAs its private 16x32 A slab
// into LDS with GLOBAL_LOAD_ASYNC_TO_LDS_B128 (next tile in flight while the
// current one feeds the matrix pipe). B fragments come straight from the
// fragment-native bf16 weight buffer in L2 (2 x global_load_b128 each).
// mode 0: outF = A@B ; mode 1: outF = res + A@B ; mode 2: outB = bf16(relu(A@B)^2)

__global__ __launch_bounds__(256) void k_gemm(const __bf16* __restrict__ A,
                                              const __bf16* __restrict__ Bt,
                                              float* __restrict__ outF,
                                              const float* __restrict__ res,
                                              __bf16* __restrict__ outB,
                                              int M, int N, int K, int mode) {
  __shared__ __bf16 As[8][2][16][40];  // per-wave double-buffered slab, padded rows
  const int tid = threadIdx.x;
  const int lane = tid & 31;
  const int wave = tid >> 5;
  const int ln = lane & 15;
  const int half = lane >> 4;
  const int row0 = blockIdx.y * 128 + wave * 16;
  const int colt0 = blockIdx.x * 4;  // first 16-col tile
  const int nkt = K >> 5;

  const unsigned base0 = (unsigned)(uintptr_t)&As[wave][0][0][0];
  const unsigned base1 = (unsigned)(uintptr_t)&As[wave][1][0][0];
  const int cr = (lane + 0) >> 2, cc = (lane + 0) & 3;        // chunk 0
  const int dr = (lane + 32) >> 2, dc = (lane + 32) & 3;      // chunk 1

  v8f zero = {};
  v8f acc[4];
#pragma unroll
  for (int s = 0; s < 4; ++s) acc[s] = zero;

  // prologue: stage K-tile 0
  async_b128(base0 + cr * 80 + cc * 16, A + (size_t)(row0 + cr) * K + cc * 8);
  async_b128(base0 + dr * 80 + dc * 16, A + (size_t)(row0 + dr) * K + dc * 8);

  for (int k0 = 0, it = 0; k0 < K; k0 += 32, ++it) {
    const int cur = it & 1;
    if (k0 + 32 < K) {
      unsigned nb = cur ? base0 : base1;
      async_b128(nb + cr * 80 + cc * 16, A + (size_t)(row0 + cr) * K + (k0 + 32) + cc * 8);
      async_b128(nb + dr * 80 + dc * 16, A + (size_t)(row0 + dr) * K + (k0 + 32) + dc * 8);
      asm volatile("s_wait_asynccnt 0x2" ::: "memory");  // current tile landed
    } else {
      asm volatile("s_wait_asynccnt 0x0" ::: "memory");
    }

    // A fragment (contiguous k = 16*half + j, matching B): two 16B chunks from this wave's slab
    const __bf16* ar = &As[wave][cur][ln][0];
    v16bf a = ld_pair(ar + half * 16, ar + half * 16 + 8);

#pragma unroll
    for (int s = 0; s < 4; ++s) {
      const __bf16* bp = Bt + (((size_t)(colt0 + s) * nkt) + (k0 >> 5)) * 512 + lane * 16;
      v16bf b = ld_pair(bp, bp + 8);
      acc[s] = __builtin_amdgcn_wmma_f32_16x16x32_bf16(false, a, false, b, (short)0,
                                                       acc[s], false, false);
    }
  }

#pragma unroll
  for (int s = 0; s < 4; ++s) {
#pragma unroll
    for (int r = 0; r < 8; ++r) {
      int row = row0 + r + half * 8;
      int col = colt0 * 16 + s * 16 + ln;
      size_t idx = (size_t)row * N + col;
      float val = acc[s][r];
      if (mode == 0) {
        outF[idx] = val;
      } else if (mode == 1) {
        outF[idx] = res[idx] + val;
      } else {
        float t = fmaxf(val, 0.0f);
        outB[idx] = (__bf16)(t * t);
      }
    }
  }
}

// ---------------------------------------------------------------- flash attention (1 wave / 16 rows / head)

__global__ __launch_bounds__(32) void k_attn(const __bf16* __restrict__ qB,
                                             const __bf16* __restrict__ kB,
                                             const __bf16* __restrict__ vT,
                                             const int* __restrict__ doc,
                                             __bf16* __restrict__ yB, int sliding) {
  __shared__ __bf16 plds[16][40];
  const int lane = threadIdx.x;
  const int ln = lane & 15;
  const int half = lane >> 4;
  const int qt = blockIdx.x;
  const int hh = blockIdx.y;
  const int kvh = hh >> 1;
  const int tlo = qt * 16;
  const int shi = tlo + 15;

  const __bf16* qp = qB + (size_t)(tlo + ln) * (NH_ * HD_) + hh * HD_;
  v16bf aq[4];
#pragma unroll
  for (int ch = 0; ch < 4; ++ch)
    aq[ch] = ld_pair(qp + ch * 32 + half * 16, qp + ch * 32 + half * 16 + 8);

  int trow[8], tdoc[8];
  float mrow[8], lrow[8];
  v8f zero = {};
  v8f acc[8];
#pragma unroll
  for (int r = 0; r < 8; ++r) {
    trow[r] = tlo + r + half * 8;
    tdoc[r] = doc[trow[r]];
    mrow[r] = -1e30f;
    lrow[r] = 0.0f;
    acc[r] = zero;
  }

  int slo = 0;
  if (sliding) {
    slo = tlo - WIN_;
    if (slo < 0) slo = 0;
  }

  for (int s0 = slo & ~31; s0 <= shi; s0 += 32) {
    v8f sc[2];
    sc[0] = zero;
    sc[1] = zero;
#pragma unroll
    for (int sub = 0; sub < 2; ++sub) {
      int key = s0 + sub * 16 + ln;
      const __bf16* kp = kB + (size_t)key * KVD_ + kvh * HD_;
#pragma unroll
      for (int ch = 0; ch < 4; ++ch) {
        const __bf16* p = kp + ch * 32 + half * 16;
        v16bf bk = ld_pair(p, p + 8);
        sc[sub] = __builtin_amdgcn_wmma_f32_16x16x32_bf16(false, aq[ch], false, bk,
                                                          (short)0, sc[sub], false, false);
      }
    }

    int sd0 = doc[s0 + ln];
    int sd1 = doc[s0 + 16 + ln];
    int scol0 = s0 + ln, scol1 = s0 + 16 + ln;
    float p0[8], p1[8], corr[8];
#pragma unroll
    for (int r = 0; r < 8; ++r) {
      int t = trow[r];
      bool ok0 = (scol0 <= t) && (sd0 == tdoc[r]) && (!sliding || (t - scol0) <= WIN_);
      bool ok1 = (scol1 <= t) && (sd1 == tdoc[r]) && (!sliding || (t - scol1) <= WIN_);
      float v0 = ok0 ? sc[0][r] * SCALE_ : -1e30f;
      float v1 = ok1 ? sc[1][r] * SCALE_ : -1e30f;
      float rm = hmax16(fmaxf(v0, v1));
      float mn = fmaxf(mrow[r], rm);
      float c = __expf(mrow[r] - mn);
      float e0 = ok0 ? __expf(v0 - mn) : 0.0f;
      float e1 = ok1 ? __expf(v1 - mn) : 0.0f;
      lrow[r] = lrow[r] * c + hsum16(e0 + e1);
      mrow[r] = mn;
      corr[r] = c;
      p0[r] = e0;
      p1[r] = e1;
    }

#pragma unroll
    for (int r = 0; r < 8; ++r) {
      plds[r + half * 8][ln] = (__bf16)p0[r];
      plds[r + half * 8][16 + ln] = (__bf16)p1[r];
    }
#pragma unroll
    for (int cs = 0; cs < 8; ++cs)
#pragma unroll
      for (int r = 0; r < 8; ++r) acc[cs][r] *= corr[r];
    __syncthreads();
    const __bf16* pr = &plds[ln][0];
    v16bf pf = ld_pair(pr + half * 16, pr + half * 16 + 8);
    __syncthreads();

#pragma unroll
    for (int cs = 0; cs < 8; ++cs) {
      const __bf16* vp = vT + ((size_t)kvh * HD_ + cs * 16 + ln) * T_ + s0 + half * 16;
      v16bf vf = ld_pair(vp, vp + 8);
      acc[cs] = __builtin_amdgcn_wmma_f32_16x16x32_bf16(false, pf, false, vf, (short)0,
                                                        acc[cs], false, false);
    }
  }

#pragma unroll
  for (int cs = 0; cs < 8; ++cs)
#pragma unroll
    for (int r = 0; r < 8; ++r) {
      float o = acc[cs][r] / lrow[r];
      yB[(size_t)trow[r] * (NH_ * HD_) + hh * HD_ + cs * 16 + ln] = (__bf16)o;
    }
}

// ---------------------------------------------------------------- launch

extern "C" void kernel_launch(void* const* d_in, const int* in_sizes, int n_in,
                              void* d_out, int out_size, void* d_ws, size_t ws_size,
                              hipStream_t stream) {
  const float* x    = (const float*)d_in[0];
  const float* ve   = (const float*)d_in[1];
  const float* Wq   = (const float*)d_in[2];
  const float* Wk   = (const float*)d_in[3];
  const float* Wv   = (const float*)d_in[4];
  const float* Wo   = (const float*)d_in[5];
  const float* Wg   = (const float*)d_in[6];
  const float* Wfc  = (const float*)d_in[7];
  const float* Wpj  = (const float*)d_in[8];
  const float* cosb = (const float*)d_in[9];
  const float* sinb = (const float*)d_in[10];
  const int* seqlens = (const int*)d_in[11];

  const size_t MB = 1024 * 1024;
  char* w = (char*)d_ws;
  float* h    = (float*)(w + 0);        // 8 MB
  float* xnF  = (float*)(w + 8 * MB);   // 8 MB
  __bf16* xnB = (__bf16*)(w + 16 * MB); // 4 MB
  float* qf   = (float*)(w + 20 * MB);  // 8 MB
  float* kf   = (float*)(w + 28 * MB);  // 4 MB
  float* vf   = (float*)(w + 32 * MB);  // 4 MB
  __bf16* qB  = (__bf16*)(w + 36 * MB); // 4 MB
  __bf16* kB  = (__bf16*)(w + 40 * MB); // 2 MB
  __bf16* vT  = (__bf16*)(w + 42 * MB); // 2 MB
  __bf16* yB  = (__bf16*)(w + 44 * MB); // 4 MB
  __bf16* mB  = (__bf16*)(w + 48 * MB); // 16 MB
  int* doc    = (int*)(w + 64 * MB);    // 8 KB
  __bf16* wqT  = (__bf16*)(w + 66 * MB); // 2 MB  (1M el)
  __bf16* wkT  = (__bf16*)(w + 68 * MB); // 1 MB
  __bf16* wvT  = (__bf16*)(w + 69 * MB); // 1 MB
  __bf16* woT  = (__bf16*)(w + 70 * MB); // 2 MB
  __bf16* wfcT = (__bf16*)(w + 72 * MB); // 8 MB
  __bf16* wpjT = (__bf16*)(w + 80 * MB); // 8 MB  -> 88 MB total

  hipMemcpyAsync(h, x, (size_t)T_ * DIM_ * sizeof(float), hipMemcpyDeviceToDevice, stream);
  k_doc<<<(T_ + 255) / 256, 256, 0, stream>>>(seqlens, doc);

  for (int l = 0; l < L_; ++l) {
    // per-layer weight conversion into fragment-native bf16 (read once, L2-hot after)
    k_convW<<<(DIM_ * DIM_) / 256, 256, 0, stream>>>(Wq + (size_t)l * DIM_ * DIM_, wqT, DIM_, DIM_);
    k_convW<<<(DIM_ * KVD_) / 256, 256, 0, stream>>>(Wk + (size_t)l * DIM_ * KVD_, wkT, DIM_, KVD_);
    k_convW<<<(DIM_ * KVD_) / 256, 256, 0, stream>>>(Wv + (size_t)l * DIM_ * KVD_, wvT, DIM_, KVD_);
    k_convW<<<(DIM_ * DIM_) / 256, 256, 0, stream>>>(Wo + (size_t)l * DIM_ * DIM_, woT, DIM_, DIM_);
    k_convW<<<(DIM_ * 4096) / 256, 256, 0, stream>>>(Wfc + (size_t)l * DIM_ * 4096, wfcT, DIM_, 4096);
    k_convW<<<(4096 * DIM_) / 256, 256, 0, stream>>>(Wpj + (size_t)l * 4096 * DIM_, wpjT, 4096, DIM_);

    k_rmsnorm<<<T_, 256, 0, stream>>>(h, xnF, xnB);

    k_gemm<<<dim3(DIM_ / 64, T_ / 128), 256, 0, stream>>>(xnB, wqT, qf, nullptr, nullptr,
                                                          T_, DIM_, DIM_, 0);
    k_gemm<<<dim3(KVD_ / 64, T_ / 128), 256, 0, stream>>>(xnB, wkT, kf, nullptr, nullptr,
                                                          T_, KVD_, DIM_, 0);
    k_gemm<<<dim3(KVD_ / 64, T_ / 128), 256, 0, stream>>>(xnB, wvT, vf, nullptr, nullptr,
                                                          T_, KVD_, DIM_, 0);

    if (l & 1)  // has_ve: l%2 == (L-1)%2 == 1
      k_gate<<<T_, 128, 0, stream>>>(xnF, Wg + (size_t)l * GATECH_ * NKV_,
                                     ve + (size_t)l * T_ * KVD_, vf);

    k_rope<<<dim3(T_, NH_ + 2 * NKV_), 128, 0, stream>>>(qf, kf, vf, cosb, sinb, qB, kB, vT);

    k_attn<<<dim3(T_ / 16, NH_), 32, 0, stream>>>(qB, kB, vT, doc, yB, (l % 4) != 3);

    k_gemm<<<dim3(DIM_ / 64, T_ / 128), 256, 0, stream>>>(yB, woT, h, h, nullptr,
                                                          T_, DIM_, DIM_, 1);

    k_rmsnorm<<<T_, 256, 0, stream>>>(h, xnF, xnB);

    k_gemm<<<dim3(4096 / 64, T_ / 128), 256, 0, stream>>>(xnB, wfcT, nullptr, nullptr, mB,
                                                          T_, 4096, DIM_, 2);

    k_gemm<<<dim3(DIM_ / 64, T_ / 128), 256, 0, stream>>>(mB, wpjT, h, h, nullptr,
                                                          T_, DIM_, 4096, 1);
  }

  hipMemcpyAsync(d_out, h, (size_t)T_ * DIM_ * sizeof(float), hipMemcpyDeviceToDevice, stream);
}
